// MultiHeadedAttention_Crosss_Relation_Decoder_42485816492441
// MI455X (gfx1250) — matrix operand order repaired and checked by reading, not compiled, not run on hardware
//
#include <hip/hip_runtime.h>
#include <math.h>

// ---------------------------------------------------------------------------
// Problem constants (from the reference)
// ---------------------------------------------------------------------------
static constexpr int BATCH = 4;
static constexpr int LQ    = 1024;
static constexpr int LK    = 1024;
static constexpr int DMODEL= 1024;
static constexpr int NHEAD = 16;
static constexpr int DK    = 64;
#define SCALE_F  (0.125f)         /* 1/sqrt(64) */
#define NEG_INF_F (-1.0e9f)

typedef __attribute__((ext_vector_type(16))) __bf16 v16bf;
typedef __attribute__((ext_vector_type(8)))  float  v8f;

// ---------------------------------------------------------------------------
// WMMA fragment loaders (gfx1250 wave32 layouts, CDNA5 ISA §7.12.2)
// A-matrix 16x32 bf16: lane 0-15 -> M=lane, K pairs {0..7,16..23};
//                      lane 16-31 -> M=lane-16, K pairs {8..15,24..31}
// ---------------------------------------------------------------------------
__device__ inline v16bf load_A16x32(const __bf16* base, int ld) {
  const int lane = threadIdx.x & 31;
  const int m  = lane & 15;
  const int kb = (lane & 16) ? 8 : 0;
  const __bf16* row = base + m * ld;
  v16bf a;
#pragma unroll
  for (int j = 0; j < 8; ++j) {
    const int k = kb + ((j < 4) ? (2 * j) : (16 + 2 * (j - 4)));
    a[2 * j]     = row[k];
    a[2 * j + 1] = row[k + 1];
  }
  return a;
}

// B-matrix 32x16 bf16 read from a K-contiguous "B^T" tile: rows = N, cols = K.
// lane 0-15 -> N=lane, K {0..7,16..23} ; lane 16-31 -> N=lane-16, K {8..15,24..31}
__device__ inline v16bf load_B32x16(const __bf16* base, int ld) {
  const int lane = threadIdx.x & 31;
  const int n  = lane & 15;
  const int kb = (lane & 16) ? 8 : 0;
  const __bf16* row = base + n * ld;
  v16bf b;
#pragma unroll
  for (int j = 0; j < 8; ++j) {
    const int k = kb + ((j < 4) ? (2 * j) : (16 + 2 * (j - 4)));
    b[2 * j]     = row[k];
    b[2 * j + 1] = row[k + 1];
  }
  return b;
}

// ---------------------------------------------------------------------------
// Half-group (16-lane) reductions for the C-layout online softmax
// ---------------------------------------------------------------------------
__device__ inline float half_max(float v) {
#pragma unroll
  for (int off = 1; off < 16; off <<= 1) v = fmaxf(v, __shfl_xor(v, off, 32));
  return v;
}
__device__ inline float half_sum(float v) {
#pragma unroll
  for (int off = 1; off < 16; off <<= 1) v += __shfl_xor(v, off, 32);
  return v;
}

// ---------------------------------------------------------------------------
// Staging: global -> registers -> LDS (bf16), software-pipelined in GEMM
// ---------------------------------------------------------------------------
template <typename T> struct StageT;
template <> struct StageT<float>  { using type = float4; };
template <> struct StageT<__bf16> { using type = uint2;  };

__device__ inline void gload4(const float* src, float4& d) {
  d = *reinterpret_cast<const float4*>(src);
}
__device__ inline void gload4(const __bf16* src, uint2& d) {
  d = *reinterpret_cast<const uint2*>(src);
}
__device__ inline void toLds4(const float4& d, __bf16* dst) {
  dst[0] = (__bf16)d.x; dst[1] = (__bf16)d.y;
  dst[2] = (__bf16)d.z; dst[3] = (__bf16)d.w;
}
__device__ inline void toLds4(const uint2& d, __bf16* dst) {
  *reinterpret_cast<uint2*>(dst) = d;
}

// direct global->LDS 4-element copy (attention tile staging)
__device__ inline void stage4(const __bf16* src, __bf16* dst) {
  *reinterpret_cast<uint2*>(dst) = *reinterpret_cast<const uint2*>(src);
}

// ---------------------------------------------------------------------------
// Kernels 1/3: Y = X(MxK) @ W^T(NxK) + bias.
//   BHLD=true : Y written bf16 in [B,H,L,DK] layout (projections)
//   BHLD=false: Y written fp32 row-major [M,N]      (output projection)
// Tile 128x128, K-step 64, register-staged double buffering: next tile's
// global loads issue before the 16-WMMA block so HBM/L2 latency hides under
// the matrix pipes. 256 threads = 8 waves in 2(M)x4(N); each wave owns 64x32.
// ---------------------------------------------------------------------------
template <typename TIN, bool BHLD>
__global__ __launch_bounds__(256)
void gemm_bias_kernel(const TIN* __restrict__ X, const float* __restrict__ W,
                      const float* __restrict__ bias, void* __restrict__ Yraw,
                      int M, int K, int N) {
  constexpr int TM = 128, TN = 128, TK = 64;
  constexpr int LDA = TK + 8, LDB = TK + 8;
  __shared__ __bf16 As[TM][LDA];
  __shared__ __bf16 Bs[TN][LDB];

  const int tid  = threadIdx.x;
  const int lane = tid & 31;
  const int w    = tid >> 5;
  const int wm   = w & 1;    // 2 waves along M (64 rows each)
  const int wn   = w >> 1;   // 4 waves along N (32 cols each)
  const int m0   = blockIdx.x * TM;
  const int n0   = blockIdx.y * TN;

  using ST = typename StageT<TIN>::type;
  ST aReg[8];
  float4 bReg[8];

  v8f acc[4][2];
#pragma unroll
  for (int mt = 0; mt < 4; ++mt)
#pragma unroll
    for (int nt = 0; nt < 2; ++nt)
#pragma unroll
      for (int r = 0; r < 8; ++r) acc[mt][nt][r] = 0.0f;

  // chunk mapping: 2048 4-elem chunks per tile; 8 per thread
  // cid = tid + 256*j ; row = cid/16 ; col = (cid%16)*4   (TK=64 -> 16 chunks/row)
  auto loadTiles = [&](int k0) {
#pragma unroll
    for (int j = 0; j < 8; ++j) {
      const int cid = tid + 256 * j;
      const int r = cid >> 4, c = (cid & 15) * 4;
      gload4(X + (size_t)(m0 + r) * K + k0 + c, aReg[j]);
      gload4(W + (size_t)(n0 + r) * K + k0 + c, bReg[j]);
    }
  };

  loadTiles(0);
  for (int k0 = 0; k0 < K; k0 += TK) {
    __syncthreads();  // previous iteration's LDS reads complete
#pragma unroll
    for (int j = 0; j < 8; ++j) {
      const int cid = tid + 256 * j;
      const int r = cid >> 4, c = (cid & 15) * 4;
      toLds4(aReg[j], &As[r][c]);
      toLds4(bReg[j], &Bs[r][c]);
    }
    __syncthreads();

    if (k0 + TK < K) loadTiles(k0 + TK);  // issue next global loads early

#pragma unroll
    for (int ks = 0; ks < 2; ++ks) {
      v16bf bfrag[2];
#pragma unroll
      for (int nt = 0; nt < 2; ++nt)
        bfrag[nt] = load_B32x16(&Bs[wn * 32 + nt * 16][ks * 32], LDB);
#pragma unroll
      for (int mt = 0; mt < 4; ++mt) {
        v16bf a = load_A16x32(&As[wm * 64 + mt * 16][ks * 32], LDA);
#pragma unroll
        for (int nt = 0; nt < 2; ++nt)
          acc[mt][nt] = __builtin_amdgcn_wmma_f32_16x16x32_bf16(
              false, a, false, bfrag[nt], (short)0, acc[mt][nt], false, false);
      }
    }
  }

  // Epilogue: bias add + store
  const int colBase = n0 + wn * 32;
  const int rowHalf = (lane & 16) ? 8 : 0;
#pragma unroll
  for (int mt = 0; mt < 4; ++mt) {
#pragma unroll
    for (int nt = 0; nt < 2; ++nt) {
      const int col = colBase + nt * 16 + (lane & 15);
      const float bv = bias[col];
      const int rowBase = m0 + wm * 64 + mt * 16 + rowHalf;
#pragma unroll
      for (int r = 0; r < 8; ++r) {
        const float vv = acc[mt][nt][r] + bv;
        const int row = rowBase + r;
        if (BHLD) {
          // row = b*L + l ; col = h*DK + dk -> [B,H,L,DK] bf16
          const int b = row >> 10, l = row & 1023;
          const int h = col >> 6,  dk = col & 63;
          ((__bf16*)Yraw)[(((size_t)(b * NHEAD + h)) * LQ + l) * DK + dk] =
              (__bf16)vv;
        } else {
          ((float*)Yraw)[(size_t)row * N + col] = vv;
        }
      }
    }
  }
}

// ---------------------------------------------------------------------------
// Kernel 2: fused dual-stream flash attention.
// Grid: B*H*(LQ/64) blocks; 128 threads = 4 waves, wave w owns q rows w*16..+16.
// ---------------------------------------------------------------------------
__device__ void stream_update(const __bf16* __restrict__ Kt, int ldk,
                              const __bf16* __restrict__ Vt, int ldv,
                              const v16bf* qa, float* m, float* l, v8f* acc,
                              const float* __restrict__ Mb,
                              __bf16* __restrict__ Pw, int lane) {
  // ---- scores: S = (Q @ K^T) * scale + mask_bias ----
  v8f s[4];
#pragma unroll
  for (int nt = 0; nt < 4; ++nt) {
#pragma unroll
    for (int r = 0; r < 8; ++r) s[nt][r] = 0.0f;
#pragma unroll
    for (int ks = 0; ks < 2; ++ks) {
      v16bf bfr = load_B32x16(Kt + (nt * 16) * ldk + ks * 32, ldk);
      s[nt] = __builtin_amdgcn_wmma_f32_16x16x32_bf16(
          false, qa[ks], false, bfr, (short)0, s[nt], false, false);
    }
  }
#pragma unroll
  for (int nt = 0; nt < 4; ++nt) {
    const float mb = Mb[nt * 16 + (lane & 15)];
#pragma unroll
    for (int r = 0; r < 8; ++r) s[nt][r] = s[nt][r] * SCALE_F + mb;
  }

  // ---- online softmax (per output row) ----
  float rmax[8], corr[8], rsum[8];
#pragma unroll
  for (int r = 0; r < 8; ++r) {
    float v = s[0][r];
#pragma unroll
    for (int nt = 1; nt < 4; ++nt) v = fmaxf(v, s[nt][r]);
    rmax[r] = half_max(v);
    const float mn = fmaxf(m[r], rmax[r]);
    corr[r] = __expf(m[r] - mn);
    m[r] = mn;
    rsum[r] = 0.0f;
  }
#pragma unroll
  for (int nt = 0; nt < 4; ++nt)
#pragma unroll
    for (int r = 0; r < 8; ++r) {
      const float p = __expf(s[nt][r] - m[r]);
      s[nt][r] = p;
      rsum[r] += p;
    }
#pragma unroll
  for (int r = 0; r < 8; ++r) {
    l[r] = l[r] * corr[r] + half_sum(rsum[r]);
#pragma unroll
    for (int dt = 0; dt < 4; ++dt) acc[dt][r] *= corr[r];
  }

  // ---- stage P (C layout -> row-major A tile in LDS) ----
  const int rowHalf = (lane & 16) ? 8 : 0;
  const int col0 = lane & 15;
#pragma unroll
  for (int nt = 0; nt < 4; ++nt)
#pragma unroll
    for (int r = 0; r < 8; ++r)
      Pw[(r + rowHalf) * 64 + nt * 16 + col0] = (__bf16)s[nt][r];
  __asm__ volatile("" ::: "memory");  // wave-private region; DS ops in-order per wave

  // ---- O += P @ V ----
  v16bf pa[2];
  pa[0] = load_A16x32(Pw, 64);
  pa[1] = load_A16x32(Pw + 32, 64);
#pragma unroll
  for (int dt = 0; dt < 4; ++dt)
#pragma unroll
    for (int ks = 0; ks < 2; ++ks) {
      v16bf bfr = load_B32x16(Vt + (dt * 16) * ldv + ks * 32, ldv);
      acc[dt] = __builtin_amdgcn_wmma_f32_16x16x32_bf16(
          false, pa[ks], false, bfr, (short)0, acc[dt], false, false);
    }
}

__global__ __launch_bounds__(128)
void attn_kernel(const __bf16* __restrict__ qp, const __bf16* __restrict__ kp,
                 const __bf16* __restrict__ vp, const __bf16* __restrict__ rkp,
                 const __bf16* __restrict__ rvp, const int* __restrict__ mask,
                 __bf16* __restrict__ X2) {
  const int blk = blockIdx.x;
  const int qt = blk & 15;
  const int h  = (blk >> 4) & 15;
  const int b  = blk >> 8;
  const int tid = threadIdx.x, lane = tid & 31, w = tid >> 5;
  const int q0 = qt * 64;
  const size_t bh = (size_t)(b * NHEAD + h);

  __shared__ __bf16 Qs[64][64];
  __shared__ __bf16 Ks[64][64];
  __shared__ __bf16 RKs[64][64];
  __shared__ __bf16 VTs[64][68];   // transposed: [dk][key], padded
  __shared__ __bf16 RVTs[64][68];
  __shared__ __bf16 Ps[4][16][64]; // per-wave P staging
  __shared__ float  Mb[64];

  // Load Q tile once ([B,H,L,DK] is dk-contiguous)
  {
    const __bf16* src = qp + (bh * LQ + q0) * DK;
    for (int i = tid; i < 64 * 64 / 4; i += 128) {
      const int r = i >> 4, c = (i & 15) * 4;
      stage4(src + r * DK + c, &Qs[r][c]);
    }
  }
  __syncthreads();
  v16bf qa[2];
  qa[0] = load_A16x32(&Qs[w * 16][0], 64);
  qa[1] = load_A16x32(&Qs[w * 16][32], 64);

  v8f acc_v[4], acc_r[4];
  float mv[8], lv[8], mr[8], lr[8];
#pragma unroll
  for (int r = 0; r < 8; ++r) {
    mv[r] = mr[r] = -3.0e38f;
    lv[r] = lr[r] = 0.0f;
#pragma unroll
    for (int dt = 0; dt < 4; ++dt) { acc_v[dt][r] = 0.0f; acc_r[dt][r] = 0.0f; }
  }

  const __bf16* kBase  = kp  + bh * LK * DK;
  const __bf16* rkBase = rkp + bh * LK * DK;
  const __bf16* vBase  = vp  + bh * LK * DK;
  const __bf16* rvBase = rvp + bh * LK * DK;

  for (int kt = 0; kt < LK / 64; ++kt) {
    const int kk0 = kt * 64;

    // Prefetch next k-tile of all four streams (global_prefetch_b8):
    // 128 lanes x 64B stride covers the 8KB tile; lands in cache before the
    // next iteration's staging loads.
    if (kt + 1 < LK / 64) {
      const size_t nOff = (size_t)(kk0 + 64) * DK + tid * 32;
      __builtin_prefetch(kBase  + nOff, 0, 3);
      __builtin_prefetch(rkBase + nOff, 0, 3);
      __builtin_prefetch(vBase  + nOff, 0, 3);
      __builtin_prefetch(rvBase + nOff, 0, 3);
    }

    __syncthreads();
    {
      const __bf16* ksrc  = kBase  + (size_t)kk0 * DK;
      const __bf16* rksrc = rkBase + (size_t)kk0 * DK;
      for (int i = tid; i < 1024; i += 128) {
        const int r = i >> 4, c = (i & 15) * 4;
        stage4(ksrc  + r * DK + c, &Ks[r][c]);
        stage4(rksrc + r * DK + c, &RKs[r][c]);
      }
      const __bf16* vsrc  = vBase  + (size_t)kk0 * DK;
      const __bf16* rvsrc = rvBase + (size_t)kk0 * DK;
      for (int i = tid; i < 1024; i += 128) {
        const int r = i >> 4, c = (i & 15) * 4;  // key r, dk c..c+3
        __bf16 t[4];
        *reinterpret_cast<uint2*>(t) =
            *reinterpret_cast<const uint2*>(vsrc + r * DK + c);
        VTs[c + 0][r] = t[0]; VTs[c + 1][r] = t[1];
        VTs[c + 2][r] = t[2]; VTs[c + 3][r] = t[3];
        *reinterpret_cast<uint2*>(t) =
            *reinterpret_cast<const uint2*>(rvsrc + r * DK + c);
        RVTs[c + 0][r] = t[0]; RVTs[c + 1][r] = t[1];
        RVTs[c + 2][r] = t[2]; RVTs[c + 3][r] = t[3];
      }
      if (tid < 64)
        Mb[tid] = (mask[b * LK + kk0 + tid] == 0) ? NEG_INF_F : 0.0f;
    }
    __syncthreads();

    stream_update(&Ks[0][0], 64, &VTs[0][0], 68, qa, mv, lv, acc_v, Mb,
                  &Ps[w][0][0], lane);
    stream_update(&RKs[0][0], 64, &RVTs[0][0], 68, qa, mr, lr, acc_r, Mb,
                  &Ps[w][0][0], lane);
  }

  // Finalize: x = acc_v/l_v + acc_r/l_r, write bf16 [B, LQ, D]
  const int rowHalf = (lane & 16) ? 8 : 0;
#pragma unroll
  for (int dt = 0; dt < 4; ++dt) {
    const int d = dt * 16 + (lane & 15);
#pragma unroll
    for (int r = 0; r < 8; ++r) {
      const int qrow = q0 + w * 16 + r + rowHalf;
      const float o = acc_v[dt][r] / lv[r] + acc_r[dt][r] / lr[r];
      X2[((size_t)b * LQ + qrow) * DMODEL + h * DK + d] = (__bf16)o;
    }
  }
}

// ---------------------------------------------------------------------------
// Launch: 5 projections -> fused attention -> output projection
// Workspace (bf16): q,k,v,rk,rv,x2 @ 8MB each = 48MB (L2-resident on MI455X)
// ---------------------------------------------------------------------------
extern "C" void kernel_launch(void* const* d_in, const int* in_sizes, int n_in,
                              void* d_out, int out_size, void* d_ws,
                              size_t ws_size, hipStream_t stream) {
  const float* query = (const float*)d_in[0];
  const float* key   = (const float*)d_in[1];
  const float* value = (const float*)d_in[2];
  const float* weak  = (const float*)d_in[3];
  const int*   mask  = (const int*)d_in[4];
  const float* Wq = (const float*)d_in[5],  *bq = (const float*)d_in[6];
  const float* Wk = (const float*)d_in[7],  *bk = (const float*)d_in[8];
  const float* Wv = (const float*)d_in[9],  *bv = (const float*)d_in[10];
  const float* Wrk = (const float*)d_in[11], *brk = (const float*)d_in[12];
  const float* Wrv = (const float*)d_in[13], *brv = (const float*)d_in[14];
  const float* Wo = (const float*)d_in[15], *bo = (const float*)d_in[16];
  float* out = (float*)d_out;

  const size_t per = (size_t)BATCH * LQ * DMODEL;  // 4,194,304 elements
  __bf16* ws  = (__bf16*)d_ws;
  __bf16* qp  = ws;
  __bf16* kp  = ws + per;
  __bf16* vp  = ws + 2 * per;
  __bf16* rkp = ws + 3 * per;
  __bf16* rvp = ws + 4 * per;
  __bf16* x2  = ws + 5 * per;

  const int M = BATCH * LQ;  // 4096
  dim3 g(M / 128, DMODEL / 128), blk(256);

  gemm_bias_kernel<float, true><<<g, blk, 0, stream>>>(query, Wq, bq, qp, M, DMODEL, DMODEL);
  gemm_bias_kernel<float, true><<<g, blk, 0, stream>>>(key,   Wk, bk, kp, M, DMODEL, DMODEL);
  gemm_bias_kernel<float, true><<<g, blk, 0, stream>>>(value, Wv, bv, vp, M, DMODEL, DMODEL);
  gemm_bias_kernel<float, true><<<g, blk, 0, stream>>>(weak,  Wrk, brk, rkp, M, DMODEL, DMODEL);
  gemm_bias_kernel<float, true><<<g, blk, 0, stream>>>(weak,  Wrv, brv, rvp, M, DMODEL, DMODEL);

  attn_kernel<<<BATCH * NHEAD * (LQ / 64), 128, 0, stream>>>(qp, kp, vp, rkp, rvp, mask, x2);

  gemm_bias_kernel<__bf16, false><<<g, blk, 0, stream>>>(x2, Wo, bo, out, M, DMODEL, DMODEL);
}
